// MipNet_44152263803439
// MI455X (gfx1250) — compile-verified, matrix operand order checked by reading, not run on hardware
//
#include <hip/hip_runtime.h>

typedef __attribute__((ext_vector_type(16))) _Float16 v16h;
typedef __attribute__((ext_vector_type(8)))  _Float16 v8h;
typedef __attribute__((ext_vector_type(8)))  float    v8f;

#define HID        512
#define MTILE      32      // sample rows per workgroup
#define ACT_PITCH  520     // 512 + 8 halfs: keeps 16B alignment, shifts banks
#define Y_PITCH    516     // 512 + 4 floats
#define LN_EPS     1e-5f

typedef union { v16h v16; v8h v8[2]; } frag_u;

// ---------------------------------------------------------------------------
// Stage hidden-layer weights fp32 -> f16 in WMMA B-fragment layout.
// Fragment index f = (l*32 + nt)*16 + kt ; within fragment: lane*16 + j halfs.
// Element (lane, j) of B fragment holds W[l][n][k] with
//   n = nt*16 + (lane & 15)
//   k = kt*32 + ( (j<8 ? j : j+8) + (lane>=16 ? 8 : 0) )
// matching the ISA 16-bit 16x32 operand layout (K pattern {0..7,16..23} /
// {8..15,24..31} split across lane halves).
// ---------------------------------------------------------------------------
__global__ void stage_weights_kernel(const float* __restrict__ Ws,
                                     _Float16* __restrict__ wsW, int total) {
  int d = blockIdx.x * blockDim.x + threadIdx.x;
  if (d >= total) return;
  int j    =  d        & 15;
  int lane = (d >> 4)  & 31;
  int kt   = (d >> 9)  & 15;
  int nt   = (d >> 13) & 31;
  int l    =  d >> 18;
  int n  = nt * 16 + (lane & 15);
  int kk = (j < 8 ? j : j + 8) + ((lane >= 16) ? 8 : 0);
  int k  = kt * 32 + kk;
  wsW[d] = (_Float16)Ws[(l * HID + n) * HID + k];
}

// LayerNorm + ReLU over active width r, Y(fp32, LDS) -> act(f16, LDS).
// 8 threads per row; intra-group reduce via 8-wide shfl_xor.
__device__ __forceinline__ void ln_relu(float (*__restrict__ Yb)[Y_PITCH],
                                        _Float16 (*__restrict__ act)[ACT_PITCH],
                                        const float* __restrict__ lw,
                                        const float* __restrict__ lb,
                                        int r, int tid) {
  const int row = tid >> 3;
  const int seg = tid & 7;
  float s = 0.f, ss = 0.f;
  for (int j = seg; j < r; j += 8) {
    float x = Yb[row][j];
    s += x; ss += x * x;
  }
  s += __shfl_xor(s, 1, 8);  ss += __shfl_xor(ss, 1, 8);
  s += __shfl_xor(s, 2, 8);  ss += __shfl_xor(ss, 2, 8);
  s += __shfl_xor(s, 4, 8);  ss += __shfl_xor(ss, 4, 8);
  const float inv  = 1.0f / (float)r;
  const float mean = s * inv;
  const float var  = ss * inv - mean * mean;
  const float rstd = rsqrtf(var + LN_EPS);
  for (int j = seg; j < r; j += 8) {
    float x = (Yb[row][j] - mean) * rstd;
    x = lw[j] * x + lb[j];
    act[row][j] = (_Float16)(x > 0.f ? x : 0.f);
  }
}

__launch_bounds__(256, 1)
__global__ void mipnet_kernel(const float* __restrict__ x_in,
                              const float* __restrict__ W_in,
                              const float* __restrict__ b_in,
                              const _Float16* __restrict__ wsW,
                              const float* __restrict__ bsv,
                              const float* __restrict__ ln_w,
                              const float* __restrict__ ln_b,
                              const float* __restrict__ W_out,
                              const float* __restrict__ b_out,
                              float* __restrict__ out) {
  __shared__ float    xin[MTILE][8];
  __shared__ _Float16 act[MTILE][ACT_PITCH];
  __shared__ float    Yb[MTILE][Y_PITCH];

  const int tid  = threadIdx.x;
  const int wave = tid >> 5;
  const int lane = tid & 31;
  const int row0 = blockIdx.x * MTILE;

  const int lhalf = lane & 15;
  const int khi   = (lane >> 4) << 3;   // 0 for lanes 0-15, 8 for lanes 16-31
  const int mhi   = khi;                // C/D row offset for upper lane half

  for (int i = tid; i < MTILE * 6; i += 256)
    xin[i / 6][i % 6] = x_in[(row0 + i / 6) * 6 + (i % 6)];
  __syncthreads();

  for (int factor = 0; factor < 3; ++factor) {
    // ------------------- input layer (K=6, scalar) -------------------
    const int r0 = (factor == 0) ? 256 : 512;
    for (int idx = tid; idx < MTILE * r0; idx += 256) {
      int m = idx / r0, n = idx - m * r0;
      float s = b_in[n];
      const float* w = &W_in[n * 6];
#pragma unroll
      for (int k = 0; k < 6; ++k) s += xin[m][k] * w[k];
      Yb[m][n] = s;
    }
    __syncthreads();
    ln_relu(Yb, act, &ln_w[0], &ln_b[0], r0, tid);
    __syncthreads();

    // ------------------- 6 hidden layers via WMMA -------------------
    int c_prev = r0;
    for (int l = 0; l < 6; ++l) {
      int r, c;
      if (factor == 0)      { r = 256; c = 256; }
      else if (factor == 2) { r = 512; c = 512; }
      else                  { if ((l & 1) == 0) { r = 256; c = 512; }
                              else              { r = 512; c = 256; } }
      const int ntiles = r >> 4;
      const int ktiles = c >> 5;

      for (int nt = wave; nt < ntiles; nt += 8) {
        const int n     = nt * 16 + lhalf;
        const float bias = bsv[l * HID + n];
        v8f acc0 = {}; v8f acc1 = {};
        const _Float16* wp = wsW + (((l * 32 + nt) * 16) << 9) + (lane << 4);
        for (int kt = 0; kt < ktiles; ++kt) {
          frag_u bfrag, a0, a1;
          bfrag.v8[0] = *(const v8h*)(wp);
          bfrag.v8[1] = *(const v8h*)(wp + 8);
          wp += 512;
          const int kbase = kt * 32 + khi;
          a0.v8[0] = *(const v8h*)&act[lhalf][kbase];
          a0.v8[1] = *(const v8h*)&act[lhalf][kbase + 16];
          a1.v8[0] = *(const v8h*)&act[16 + lhalf][kbase];
          a1.v8[1] = *(const v8h*)&act[16 + lhalf][kbase + 16];
          acc0 = __builtin_amdgcn_wmma_f32_16x16x32_f16(
              false, a0.v16, false, bfrag.v16, (short)0, acc0, false, false);
          acc1 = __builtin_amdgcn_wmma_f32_16x16x32_f16(
              false, a1.v16, false, bfrag.v16, (short)0, acc1, false, false);
        }
#pragma unroll
        for (int v = 0; v < 8; ++v) {
          Yb[v + mhi][n]      = acc0[v] + bias;
          Yb[16 + v + mhi][n] = acc1[v] + bias;
        }
      }
      __syncthreads();
      ln_relu(Yb, act, &ln_w[(l + 1) * HID], &ln_b[(l + 1) * HID], r, tid);
      __syncthreads();
      c_prev = r;
    }

    // ------------------- output layer (3 outs, scalar) -------------------
    for (int t = tid; t < MTILE * 3; t += 256) {
      int m = t / 3, o = t - m * 3;
      float s = b_out[o];
      const float* w = &W_out[o * HID];
      for (int k = 0; k < c_prev; ++k)
        s += (float)act[m][k] * w[k];
      out[((row0 + m) * 3 + factor) * 3 + o] = s;
    }
    __syncthreads();
  }
}

extern "C" void kernel_launch(void* const* d_in, const int* in_sizes, int n_in,
                              void* d_out, int out_size, void* d_ws, size_t ws_size,
                              hipStream_t stream) {
  (void)n_in; (void)out_size; (void)ws_size;
  const float* x_in  = (const float*)d_in[0];
  const float* W_in  = (const float*)d_in[1];
  const float* b_in  = (const float*)d_in[2];
  const float* Ws    = (const float*)d_in[3];
  const float* bsv   = (const float*)d_in[4];
  const float* lnw   = (const float*)d_in[5];
  const float* lnb   = (const float*)d_in[6];
  const float* W_out = (const float*)d_in[7];
  const float* b_out = (const float*)d_in[8];
  float* out = (float*)d_out;
  _Float16* wsW = (_Float16*)d_ws;

  const int N = in_sizes[0] / 6;
  const int total = 6 * 32 * 16 * 32 * 16;   // 6 layers of staged fragments
  stage_weights_kernel<<<(total + 255) / 256, 256, 0, stream>>>(Ws, wsW, total);
  mipnet_kernel<<<N / MTILE, 256, 0, stream>>>(x_in, W_in, b_in, wsW, bsv,
                                               lnw, lnb, W_out, b_out, out);
}
